// Attention_53635551593134
// MI455X (gfx1250) — compile-verified, matrix operand order checked by reading, not run on hardware
//
#include <hip/hip_runtime.h>

#define BB 4
#define NN 4096
#define UU 512
#define BN (BB*NN)

typedef __attribute__((ext_vector_type(16))) __bf16 v16bf;
typedef __attribute__((ext_vector_type(8)))  __bf16 v8bf;
typedef __attribute__((ext_vector_type(8)))  float  v8f;
typedef __attribute__((ext_vector_type(4)))  unsigned int v4u;
typedef __attribute__((ext_vector_type(8)))  int v8i;
typedef __attribute__((ext_vector_type(4)))  int v4i;

__device__ __forceinline__ v8f wmma_bf16(v16bf a, v16bf b, v8f c) {
  return __builtin_amdgcn_wmma_f32_16x16x32_bf16(false, a, false, b, (short)0, c,
                                                 false, false);
}

// ISA 16-bit A-layout (16x32 MxK): lane L holds row M=L%16,
// element i -> K = (i<8 ? i : i+8) + (L>=16 ? 8 : 0). B (32x16 KxN) mirrors it
// with N = L%16. Caller pre-offsets the pointer by kbase = (L&16)?8:0.
// Operands are stored with the contraction dim contiguous -> two 16B loads.
__device__ __forceinline__ v16bf frag_bf16(const __bf16* p) {
  v8bf lo = *(const v8bf*)(p);
  v8bf hi = *(const v8bf*)(p + 16);
  return __builtin_shufflevector(lo, hi, 0, 1, 2, 3, 4, 5, 6, 7,
                                 8, 9, 10, 11, 12, 13, 14, 15);
}
__device__ __forceinline__ v16bf frag_f32(const float* p) {
  float4 a0 = *(const float4*)(p);
  float4 a1 = *(const float4*)(p + 4);
  float4 b0 = *(const float4*)(p + 16);
  float4 b1 = *(const float4*)(p + 20);
  v16bf f;
  f[0] = (__bf16)a0.x;  f[1] = (__bf16)a0.y;  f[2] = (__bf16)a0.z;  f[3] = (__bf16)a0.w;
  f[4] = (__bf16)a1.x;  f[5] = (__bf16)a1.y;  f[6] = (__bf16)a1.z;  f[7] = (__bf16)a1.w;
  f[8] = (__bf16)b0.x;  f[9] = (__bf16)b0.y;  f[10] = (__bf16)b0.z; f[11] = (__bf16)b0.w;
  f[12] = (__bf16)b1.x; f[13] = (__bf16)b1.y; f[14] = (__bf16)b1.z; f[15] = (__bf16)b1.w;
  return f;
}

// TDM: async DMA of a 32x512 bf16 tile (row stride 512) global -> LDS.
// D# per CDNA5 ISA ch.8; 6-arg builtin on this toolchain.
__device__ __forceinline__ void tdm_load_k_tile(const __bf16* gsrc, unsigned lds_off) {
  unsigned long long ga = (unsigned long long)(uintptr_t)gsrc;
  v4u g0;
  g0[0] = 1u;                                    // count=1, user descriptor
  g0[1] = lds_off;                               // lds_addr (bytes)
  g0[2] = (unsigned)ga;                          // global_addr[31:0]
  g0[3] = (unsigned)((ga >> 32) & 0x1FFFFFFu)    // global_addr[56:32]
          | (2u << 30);                          // type=2 ("image")
  v8i g1;
  g1[0] = 0x00010000;                            // data_size=1 (2 bytes)
  g1[1] = (int)(512u << 16);                     // tensor_dim0[15:0]=512
  g1[2] = (int)(4096u << 16);                    // tensor_dim1[15:0]=4096
  g1[3] = (int)(512u << 16);                     // tile_dim0=512
  g1[4] = 32;                                    // tile_dim1=32
  g1[5] = 512;                                   // tensor_dim0_stride=512
  g1[6] = 0;
  g1[7] = 0;
  v4i z4 = {0, 0, 0, 0};
  v8i z8 = {0, 0, 0, 0, 0, 0, 0, 0};
  __builtin_amdgcn_tensor_load_to_lds(g0, g1, z4, z4, z8, 0);
}

// ---------------------------------------------------------------- kernel 0
__global__ void cvt_weights_t(const float* __restrict__ wq, const float* __restrict__ wk,
                              const float* __restrict__ wv, const float* __restrict__ wp,
                              __bf16* __restrict__ dq, __bf16* __restrict__ dk,
                              __bf16* __restrict__ dv, __bf16* __restrict__ dp) {
  const int idx = blockIdx.x * blockDim.x + threadIdx.x;   // coalesced read
  const int k = idx / UU, n = idx % UU;
  const size_t t = (size_t)n * UU + k;                     // transposed write
  dq[t] = (__bf16)wq[idx];
  dk[t] = (__bf16)wk[idx];
  dv[t] = (__bf16)wv[idx];
  dp[t] = (__bf16)wp[idx];
}

// ---------------------------------------------------------------- kernel 1
// Fused Q/K/V projection; 16x32 output tile of all three per wave.
// Manual software pipeline: fragments for step k+1 are loaded into fresh
// values before step k's wmma chain, then rotated.
__global__ void __launch_bounds__(256) qkv_proj(
    const float* __restrict__ x,
    const __bf16* __restrict__ WqT, const __bf16* __restrict__ WkT,
    const __bf16* __restrict__ WvT,
    const float* __restrict__ bqp, const float* __restrict__ bkp,
    const float* __restrict__ bvp,
    __bf16* __restrict__ Q, __bf16* __restrict__ K, __bf16* __restrict__ VT) {
  const int lane = threadIdx.x & 31;
  const int wave = threadIdx.x >> 5;
  const int tile = blockIdx.x * 8 + wave;       // 16384 tiles
  const int colPairs = UU / 32;                 // 16
  const int rowTile = tile / colPairs;
  const int colPair = tile - rowTile * colPairs;
  const int row0 = rowTile * 16, col0 = colPair * 32;
  const int kbase = (lane & 16) ? 8 : 0;
  const int r = lane & 15;
  const int mh = (lane >> 4) << 3;

  v8f cq0 = {}, cq1 = {}, ck0 = {}, ck1 = {}, cv0 = {}, cv1 = {};
  const float*  xrow = x + (size_t)(row0 + r) * UU + kbase;
  const __bf16* wp0 = WqT + (size_t)(col0 + r) * UU + kbase;
  const __bf16* wp1 = WqT + (size_t)(col0 + 16 + r) * UU + kbase;
  const __bf16* wp2 = WkT + (size_t)(col0 + r) * UU + kbase;
  const __bf16* wp3 = WkT + (size_t)(col0 + 16 + r) * UU + kbase;
  const __bf16* wp4 = WvT + (size_t)(col0 + r) * UU + kbase;
  const __bf16* wp5 = WvT + (size_t)(col0 + 16 + r) * UU + kbase;

  // pipeline prologue: stage 0 fragments
  v16bf a  = frag_f32(xrow);
  v16bf w0 = frag_bf16(wp0), w1 = frag_bf16(wp1), w2 = frag_bf16(wp2);
  v16bf w3 = frag_bf16(wp3), w4 = frag_bf16(wp4), w5 = frag_bf16(wp5);
#pragma unroll 4
  for (int k0 = 0; k0 < UU; k0 += 32) {
    v16bf an, n0, n1, n2, n3, n4, n5;
    const int k1 = k0 + 32;
    if (k1 < UU) {                               // load next step's fragments
      an = frag_f32(xrow + k1);
      n0 = frag_bf16(wp0 + k1);  n1 = frag_bf16(wp1 + k1);
      n2 = frag_bf16(wp2 + k1);  n3 = frag_bf16(wp3 + k1);
      n4 = frag_bf16(wp4 + k1);  n5 = frag_bf16(wp5 + k1);
    }
    cq0 = wmma_bf16(a, w0, cq0);
    cq1 = wmma_bf16(a, w1, cq1);
    ck0 = wmma_bf16(a, w2, ck0);
    ck1 = wmma_bf16(a, w3, ck1);
    cv0 = wmma_bf16(a, w4, cv0);
    cv1 = wmma_bf16(a, w5, cv1);
    a = an; w0 = n0; w1 = n1; w2 = n2; w3 = n3; w4 = n4; w5 = n5;
  }
  const float scale = rsqrtf((float)UU);
  const int b = row0 >> 12;                     // row0 / NN
#pragma unroll
  for (int v = 0; v < 8; ++v) {
    const int M = v + mh;
    const int nrow = (row0 + M) & (NN - 1);
    const size_t idx = (size_t)(row0 + M) * UU;
#pragma unroll
    for (int h = 0; h < 2; ++h) {
      const int Nc = col0 + 16 * h + r;
      const float q = (h ? cq1[v] : cq0[v]) + bqp[Nc];
      const float k = (h ? ck1[v] : ck0[v]) + bkp[Nc];
      const float vv = (h ? cv1[v] : cv0[v]) + bvp[Nc];
      Q[idx + Nc] = (__bf16)(q * scale);
      K[idx + Nc] = (__bf16)k;
      VT[((size_t)b * UU + Nc) * NN + nrow] = (__bf16)vv;   // transposed V
    }
  }
}

// ---------------------------------------------------------------- kernel 2
// Flash attention: 16 query rows/block, 4 waves split d (128 each).
// K tiles are TDM double-buffered in LDS; wave 0 drives the DMA.
__global__ void __launch_bounds__(128) flash_attn(
    const __bf16* __restrict__ Q, const __bf16* __restrict__ K,
    const __bf16* __restrict__ VT, __bf16* __restrict__ C) {
  __shared__ alignas(16) __bf16 sK[2][32][UU];    // 64 KB: TDM K tiles
  __shared__ float sS[4][16][32];                 // per-wave partial S
  __shared__ alignas(16) __bf16 sP[4][16][32];    // P for A-frag repack

  const int lane = threadIdx.x & 31;
  const int wave = threadIdx.x >> 5;              // 0..3 -> d-slice
  const int b  = blockIdx.y;
  const int q0 = blockIdx.x * 16;
  const int ds0 = wave * 128;
  const int kbase = (lane & 16) ? 8 : 0;
  const int r  = lane & 15;
  const int mh = (lane >> 4) << 3;

  const __bf16* Qb = Q + (size_t)b * NN * UU;
  const __bf16* Kb = K + (size_t)b * NN * UU;
  const __bf16* Vt = VT + (size_t)b * UU * NN;    // [d][n]

  typedef __attribute__((address_space(3))) __bf16 lds_bf16;
  const unsigned sK0 = (unsigned)(unsigned long long)(lds_bf16*)&sK[0][0][0];
  const unsigned sK1 = (unsigned)(unsigned long long)(lds_bf16*)&sK[1][0][0];

  v16bf qf[4];
#pragma unroll
  for (int kk = 0; kk < 4; ++kk)
    qf[kk] = frag_bf16(Qb + (size_t)(q0 + r) * UU + ds0 + kk * 32 + kbase);

  v8f o[8] = {};
  float m[8], l[8];
#pragma unroll
  for (int v = 0; v < 8; ++v) { m[v] = -3.0e38f; l[v] = 0.f; }

  // prologue: stage key tile 0
  if (wave == 0) {
    tdm_load_k_tile(Kb, sK0);
    __builtin_amdgcn_s_wait_tensorcnt(0);
  }
  __syncthreads();

  for (int kb = 0; kb < NN; kb += 32) {
    const int buf = (kb >> 5) & 1;
    // kick DMA for the next key tile while we compute on this one
    if (wave == 0 && kb + 32 < NN)
      tdm_load_k_tile(Kb + (size_t)(kb + 32) * UU, buf ? sK0 : sK1);

    // partial S over this wave's d-slice from LDS (ds_load_b128 fragments)
    v8f s0 = {}, s1 = {};
#pragma unroll
    for (int kk = 0; kk < 4; ++kk) {
      v16bf b0 = frag_bf16(&sK[buf][r][ds0 + kk * 32 + kbase]);
      v16bf b1 = frag_bf16(&sK[buf][16 + r][ds0 + kk * 32 + kbase]);
      s0 = wmma_bf16(qf[kk], b0, s0);
      s1 = wmma_bf16(qf[kk], b1, s1);
    }
    if (kb + 32 < NN)
      __builtin_prefetch(Vt + (size_t)(ds0 + r) * NN + kb + 32, 0, 1);

#pragma unroll
    for (int v = 0; v < 8; ++v) {
      sS[wave][v + mh][r]      = s0[v];
      sS[wave][v + mh][16 + r] = s1[v];
    }
    __syncthreads();
    float fs0[8], fs1[8];
#pragma unroll
    for (int v = 0; v < 8; ++v) {
      float a0 = 0.f, a1 = 0.f;
#pragma unroll
      for (int w = 0; w < 4; ++w) { a0 += sS[w][v + mh][r]; a1 += sS[w][v + mh][16 + r]; }
      fs0[v] = a0; fs1[v] = a1;
    }
    // online softmax; row v+mh lives across one 16-lane half -> xor reduce
#pragma unroll
    for (int v = 0; v < 8; ++v) {
      float mt = fmaxf(fs0[v], fs1[v]);
#pragma unroll
      for (int off = 1; off < 16; off <<= 1) mt = fmaxf(mt, __shfl_xor(mt, off, 32));
      const float mnew = fmaxf(m[v], mt);
      const float p0 = __expf(fs0[v] - mnew);
      const float p1 = __expf(fs1[v] - mnew);
      float rs = p0 + p1;
#pragma unroll
      for (int off = 1; off < 16; off <<= 1) rs += __shfl_xor(rs, off, 32);
      const float corr = __expf(m[v] - mnew);
      l[v] = l[v] * corr + rs;
      m[v] = mnew;
#pragma unroll
      for (int j = 0; j < 8; ++j) o[j][v] *= corr;
      sP[wave][v + mh][r]      = (__bf16)p0;
      sP[wave][v + mh][16 + r] = (__bf16)p1;
    }
    __syncthreads();
    // repack P (C-layout) into an A-fragment; 16B-aligned LDS vector reads
    v16bf pf = frag_bf16(&sP[wave][r][0] + kbase);
    // O[:, ds0+16j] += P * V^T; V frags loaded in groups of 4 ahead of MMAs
#pragma unroll
    for (int g = 0; g < 2; ++g) {
      v16bf f0 = frag_bf16(Vt + (size_t)(ds0 + (4 * g + 0) * 16 + r) * NN + kb + kbase);
      v16bf f1 = frag_bf16(Vt + (size_t)(ds0 + (4 * g + 1) * 16 + r) * NN + kb + kbase);
      v16bf f2 = frag_bf16(Vt + (size_t)(ds0 + (4 * g + 2) * 16 + r) * NN + kb + kbase);
      v16bf f3 = frag_bf16(Vt + (size_t)(ds0 + (4 * g + 3) * 16 + r) * NN + kb + kbase);
      o[4 * g + 0] = wmma_bf16(pf, f0, o[4 * g + 0]);
      o[4 * g + 1] = wmma_bf16(pf, f1, o[4 * g + 1]);
      o[4 * g + 2] = wmma_bf16(pf, f2, o[4 * g + 2]);
      o[4 * g + 3] = wmma_bf16(pf, f3, o[4 * g + 3]);
    }
    // publish next K tile (and retire this iteration's sS/sP reads)
    if (wave == 0 && kb + 32 < NN)
      __builtin_amdgcn_s_wait_tensorcnt(0);
    __syncthreads();
  }
#pragma unroll
  for (int v = 0; v < 8; ++v) {
    const float inv = 1.f / l[v];
    const size_t row = (size_t)b * NN + q0 + v + mh;
#pragma unroll
    for (int j = 0; j < 8; ++j)
      C[row * UU + ds0 + j * 16 + r] = (__bf16)(o[j][v] * inv);
  }
}

// ---------------------------------------------------------------- kernel 3
// out = x + ctx @ Wp + bp (fp32 out); 16x64 tile per wave, software-pipelined
__global__ void __launch_bounds__(256) out_proj(
    const float* __restrict__ x, const __bf16* __restrict__ Cb,
    const __bf16* __restrict__ WpT, const float* __restrict__ bpp,
    float* __restrict__ out) {
  const int lane = threadIdx.x & 31;
  const int wave = threadIdx.x >> 5;
  const int tile = blockIdx.x * 8 + wave;       // 8192 tiles
  const int colQuads = UU / 64;                 // 8
  const int rowTile = tile / colQuads;
  const int colQuad = tile - rowTile * colQuads;
  const int row0 = rowTile * 16, col0 = colQuad * 64;
  const int kbase = (lane & 16) ? 8 : 0;
  const int r = lane & 15;
  const int mh = (lane >> 4) << 3;

  v8f acc[4] = {};
  const __bf16* arow = Cb + (size_t)(row0 + r) * UU + kbase;
  const __bf16* w0p = WpT + (size_t)(col0 + r) * UU + kbase;
  const __bf16* w1p = WpT + (size_t)(col0 + 16 + r) * UU + kbase;
  const __bf16* w2p = WpT + (size_t)(col0 + 32 + r) * UU + kbase;
  const __bf16* w3p = WpT + (size_t)(col0 + 48 + r) * UU + kbase;

  v16bf a  = frag_bf16(arow);
  v16bf w0 = frag_bf16(w0p), w1 = frag_bf16(w1p);
  v16bf w2 = frag_bf16(w2p), w3 = frag_bf16(w3p);
#pragma unroll 4
  for (int k0 = 0; k0 < UU; k0 += 32) {
    v16bf an, n0, n1, n2, n3;
    const int k1 = k0 + 32;
    if (k1 < UU) {
      an = frag_bf16(arow + k1);
      n0 = frag_bf16(w0p + k1);  n1 = frag_bf16(w1p + k1);
      n2 = frag_bf16(w2p + k1);  n3 = frag_bf16(w3p + k1);
    }
    acc[0] = wmma_bf16(a, w0, acc[0]);
    acc[1] = wmma_bf16(a, w1, acc[1]);
    acc[2] = wmma_bf16(a, w2, acc[2]);
    acc[3] = wmma_bf16(a, w3, acc[3]);
    a = an; w0 = n0; w1 = n1; w2 = n2; w3 = n3;
  }
#pragma unroll
  for (int t = 0; t < 4; ++t) {
    const int Nc = col0 + t * 16 + r;
    const float bias = bpp[Nc];
#pragma unroll
    for (int v = 0; v < 8; ++v) {
      const size_t idx = (size_t)(row0 + v + mh) * UU + Nc;
      out[idx] = x[idx] + acc[t][v] + bias;
    }
  }
}

// ---------------------------------------------------------------- launch
extern "C" void kernel_launch(void* const* d_in, const int* in_sizes, int n_in,
                              void* d_out, int out_size, void* d_ws, size_t ws_size,
                              hipStream_t stream) {
  const float* x  = (const float*)d_in[0];
  const float* Wq = (const float*)d_in[1];
  const float* bq = (const float*)d_in[2];
  const float* Wk = (const float*)d_in[3];
  const float* bk = (const float*)d_in[4];
  const float* Wv = (const float*)d_in[5];
  const float* bv = (const float*)d_in[6];
  const float* Wp = (const float*)d_in[7];
  const float* bp = (const float*)d_in[8];
  float* out = (float*)d_out;

  char* ws = (char*)d_ws;
  const size_t wbytes = (size_t)UU * UU * sizeof(__bf16);        // 512 KB
  const size_t tbytes = (size_t)BN * UU * sizeof(__bf16);        // 16 MB
  __bf16* WqT = (__bf16*)(ws);
  __bf16* WkT = (__bf16*)(ws + wbytes);
  __bf16* WvT = (__bf16*)(ws + 2 * wbytes);
  __bf16* WpT = (__bf16*)(ws + 3 * wbytes);
  __bf16* Qb  = (__bf16*)(ws + 4 * wbytes);
  __bf16* Kb  = (__bf16*)(ws + 4 * wbytes + tbytes);
  __bf16* VTb = (__bf16*)(ws + 4 * wbytes + 2 * tbytes);
  __bf16* Ctx = (__bf16*)(ws + 4 * wbytes + 3 * tbytes);

  cvt_weights_t<<<(UU * UU) / 256, 256, 0, stream>>>(Wq, Wk, Wv, Wp, WqT, WkT, WvT, WpT);
  qkv_proj<<<2048, 256, 0, stream>>>(x, WqT, WkT, WvT, bq, bk, bv, Qb, Kb, VTb);
  dim3 fgrid(NN / 16, BB);
  flash_attn<<<fgrid, 128, 0, stream>>>(Qb, Kb, VTb, Ctx);
  out_proj<<<1024, 256, 0, stream>>>(x, Ctx, WpT, bp, out);
}
